// LSTM_45140106281058
// MI455X (gfx1250) — compile-verified, matrix operand order checked by reading, not run on hardware
//
#include <hip/hip_runtime.h>
#include <hip/hip_bf16.h>

typedef _Float16 v16h __attribute__((ext_vector_type(16)));
typedef _Float16 v8h  __attribute__((ext_vector_type(8)));
typedef float    v8f  __attribute__((ext_vector_type(8)));

#define HID   128
#define SEQ   256
#define BTILE 16
#define NCLS  10
#define LDSK  136   // padded K stride in halves: 272B row -> 4-bank shift/row, conflict-free b128

__device__ __forceinline__ float fast_sigmoid(float x) {
    // sigma(x) = 1 / (1 + 2^(-x*log2(e)))   -> v_exp_f32 + v_rcp_f32
    float e = __builtin_amdgcn_exp2f(x * -1.4426950408889634f);
    return __builtin_amdgcn_rcpf(1.0f + e);
}

#if __has_builtin(__builtin_amdgcn_tanhf)
__device__ __forceinline__ float fast_tanh(float x) { return __builtin_amdgcn_tanhf(x); }
#else
__device__ __forceinline__ float fast_tanh(float x) { return 2.0f * fast_sigmoid(2.0f * x) - 1.0f; }
#endif

__global__ __launch_bounds__(256)
void lstm_wmma_kernel(const float* __restrict__ x,
                      const float* __restrict__ wgx, const float* __restrict__ wgh, const float* __restrict__ bg,
                      const float* __restrict__ wix, const float* __restrict__ wih, const float* __restrict__ bi,
                      const float* __restrict__ wfx, const float* __restrict__ wfh, const float* __restrict__ bf,
                      const float* __restrict__ wox, const float* __restrict__ woh, const float* __restrict__ bo,
                      const float* __restrict__ wp,  const float* __restrict__ bp,
                      float* __restrict__ out_h, float* __restrict__ out_y)
{
    __shared__ float xs[BTILE][SEQ];                       // 16 KB staged input tile
    __shared__ __align__(16) _Float16 hb[2][BTILE][LDSK];  // ping-pong h (f16), padded

    const int tid    = threadIdx.x;
    const int wave   = tid >> 5;
    const int lane   = tid & 31;
    const int laneHi = lane >> 4;     // 0 or 1 (half-wave)
    const int lane16 = lane & 15;
    const int b0     = blockIdx.x * BTILE;
    const int n0     = wave * 16;          // this wave's hidden-column slice
    const int nLane  = n0 + lane16;        // N index owned by this lane (B frag / C frag)
    const int koff   = laneHi * 8;         // A/B half-wave K offset

    // ---- stage x tile (coalesced) and zero h buffer 0 ----
    for (int i = tid; i < BTILE * SEQ; i += 256)
        xs[i / SEQ][i % SEQ] = x[(size_t)(b0 + i / SEQ) * SEQ + (i % SEQ)];
    for (int i = tid; i < BTILE * LDSK; i += 256)
        (&hb[0][0][0])[i] = (_Float16)0.0f;

    // ---- per-lane input-projection weights & biases (input_dim == 1) ----
    const float gwx = wgx[nLane], gbb = bg[nLane];
    const float iwx = wix[nLane], ibb = bi[nLane];
    const float fwx = wfx[nLane], fbb = bf[nLane];
    const float owx = wox[nLane], obb = bo[nLane];

    // ---- load recurrent weights once into B-fragment registers (f32 -> f16) ----
    // B element (K=k, N=n) = W[n][k]  (out = h @ W^T). Layout mirrors A:
    // lanes 0-15: N=lane, K {0..7, 16..23}; lanes 16-31: N=lane-16, K {8..15, 24..31}
    const float* Wptr[4] = {wgh, wih, wfh, woh};
    v16h Bfrag[4][4];
    #pragma unroll
    for (int g = 0; g < 4; ++g) {
        const float* W = Wptr[g];
        #pragma unroll
        for (int ks = 0; ks < 4; ++ks) {
            v16h b;
            #pragma unroll
            for (int i = 0; i < 8; ++i) {
                b[i]     = (_Float16)W[(size_t)nLane * HID + 32 * ks + koff + i];
                b[i + 8] = (_Float16)W[(size_t)nLane * HID + 32 * ks + 16 + koff + i];
            }
            Bfrag[g][ks] = b;
        }
    }

    float cst[8], hreg[8];
    #pragma unroll
    for (int j = 0; j < 8; ++j) { cst[j] = 0.0f; hreg[j] = 0.0f; }

    __syncthreads();   // xs + hb[0] ready

    // ---- seed accumulators for t=0 with x-projection + bias ----
    v8f cg, ci, cf, co;
    #pragma unroll
    for (int j = 0; j < 8; ++j) {
        float xv = xs[laneHi * 8 + j][0];
        cg[j] = xv * gwx + gbb;
        ci[j] = xv * iwx + ibb;
        cf[j] = xv * fwx + fbb;
        co[j] = xv * owx + obb;
    }

    int p = 0;
    for (int t = 0; t < SEQ; ++t) {
        // ---- A fragments of h (16x32 f16 per K-step) from LDS, 2x b128 per kstep ----
        v16h A[4];
        #pragma unroll
        for (int ks = 0; ks < 4; ++ks) {
            const _Float16* base = &hb[p][lane16][32 * ks + koff];
            v8h lo = *(const v8h*)(base);        // K = 32ks+koff .. +7
            v8h hi = *(const v8h*)(base + 16);   // K = 32ks+16+koff .. +7
            A[ks] = __builtin_shufflevector(lo, hi, 0,1,2,3,4,5,6,7,8,9,10,11,12,13,14,15);
        }

        // ---- 4 gates x 4 K-steps of v_wmma_f32_16x16x32_f16 ----
        #pragma unroll
        for (int ks = 0; ks < 4; ++ks) {
            cg = __builtin_amdgcn_wmma_f32_16x16x32_f16(false, A[ks], false, Bfrag[0][ks], (short)0, cg, false, false);
            ci = __builtin_amdgcn_wmma_f32_16x16x32_f16(false, A[ks], false, Bfrag[1][ks], (short)0, ci, false, false);
            cf = __builtin_amdgcn_wmma_f32_16x16x32_f16(false, A[ks], false, Bfrag[2][ks], (short)0, cf, false, false);
            co = __builtin_amdgcn_wmma_f32_16x16x32_f16(false, A[ks], false, Bfrag[3][ks], (short)0, co, false, false);
        }

        // ---- elementwise cell update (f32 state in VGPRs) ----
        #pragma unroll
        for (int j = 0; j < 8; ++j) {
            float g  = fast_tanh(cg[j]);
            float ig = fast_sigmoid(ci[j]);
            float fg = fast_sigmoid(cf[j]);
            float og = fast_sigmoid(co[j]);
            float cn = g * ig + cst[j] * fg;
            cst[j]  = cn;
            hreg[j] = fast_tanh(cn) * og;
        }

        // ---- publish h (f16) to the other buffer ----
        #pragma unroll
        for (int j = 0; j < 8; ++j)
            hb[p ^ 1][laneHi * 8 + j][nLane] = (_Float16)hreg[j];

        // ---- split workgroup barrier: signal once our LDS stores landed ----
        asm volatile("s_wait_dscnt 0x0\n\ts_barrier_signal -1" ::: "memory");

        // ---- hide barrier latency: seed accumulators for t+1 (independent of hb) ----
        {
            int tn = (t + 1 < SEQ) ? (t + 1) : (SEQ - 1);   // last iter: harmless dummy
            #pragma unroll
            for (int j = 0; j < 8; ++j) {
                float xv = xs[laneHi * 8 + j][tn];
                cg[j] = xv * gwx + gbb;
                ci[j] = xv * iwx + ibb;
                cf[j] = xv * fwx + fbb;
                co[j] = xv * owx + obb;
            }
        }

        asm volatile("s_barrier_wait -1" ::: "memory");
        p ^= 1;
    }

    // ---- final hidden state (f32, from registers) ----
    #pragma unroll
    for (int j = 0; j < 8; ++j)
        out_h[(size_t)(b0 + laneHi * 8 + j) * HID + nLane] = hreg[j];

    // ---- classifier projection: y = h @ wp^T + bp  (16 x 10 per WG, trivial) ----
    for (int idx = tid; idx < BTILE * NCLS; idx += 256) {
        int r = idx / NCLS, cls = idx % NCLS;
        float acc = bp[cls];
        #pragma unroll 4
        for (int k = 0; k < HID; ++k)
            acc += (float)hb[p][r][k] * wp[cls * HID + k];
        out_y[(size_t)(b0 + r) * NCLS + cls] = acc;
    }
}

extern "C" void kernel_launch(void* const* d_in, const int* in_sizes, int n_in,
                              void* d_out, int out_size, void* d_ws, size_t ws_size,
                              hipStream_t stream) {
    (void)in_sizes; (void)n_in; (void)out_size; (void)d_ws; (void)ws_size;
    const float* x   = (const float*)d_in[0];
    const float* wgx = (const float*)d_in[1];
    const float* wgh = (const float*)d_in[2];
    const float* bg  = (const float*)d_in[3];
    const float* wix = (const float*)d_in[4];
    const float* wih = (const float*)d_in[5];
    const float* bi  = (const float*)d_in[6];
    const float* wfx = (const float*)d_in[7];
    const float* wfh = (const float*)d_in[8];
    const float* bf  = (const float*)d_in[9];
    const float* wox = (const float*)d_in[10];
    const float* woh = (const float*)d_in[11];
    const float* bo  = (const float*)d_in[12];
    const float* wp  = (const float*)d_in[13];
    const float* bp  = (const float*)d_in[14];

    float* out_h = (float*)d_out;                    // (2048, 128)
    float* out_y = out_h + (size_t)2048 * HID;       // (2048, 10)

    dim3 grid(2048 / BTILE);   // 128 workgroups, each owns 16 batch rows
    dim3 block(256);           // 8 waves; each wave owns 16 hidden columns
    lstm_wmma_kernel<<<grid, block, 0, stream>>>(x, wgx, wgh, bg, wix, wih, bi,
                                                 wfx, wfh, bf, wox, woh, bo,
                                                 wp, bp, out_h, out_y);
}
